// CDCN_14628658610713
// MI455X (gfx1250) — matrix operand order repaired and checked by reading, not compiled, not run on hardware
//
#include <hip/hip_runtime.h>
#include <math.h>

typedef __attribute__((ext_vector_type(16))) _Float16 v16h;
typedef __attribute__((ext_vector_type(8)))  _Float16 v8h;
typedef __attribute__((ext_vector_type(8)))  float    v8f;
typedef int v4i __attribute__((vector_size(16)));
typedef __attribute__((address_space(1))) v4i* gas_v4i_ptr;
typedef __attribute__((address_space(3))) v4i* las_v4i_ptr;

#define Bsz   1024
#define Hs    28
#define Ws    28
#define HW    784        // 28*28 = 49*16
#define TILES_PER_IMG 49
#define TOTAL_TILES   (Bsz * TILES_PER_IMG)   // 50176
#define EPSV  1e-5f

// ---------------------------------------------------------------------------
// Kernel 1: pack OIHW f32 weights into WMMA B-fragment order, f16.
// K ordering is TAP-MAJOR: k = tap*Cin + c  (matches NHWC A gather).
// dword d -> j = d%8, lane = (d/8)%32, nt = (d/256)%NT, chunk = d/(256*NT)
// lane: n_loc = lane&15, halfB = lane>>4 ; k_local = 16*halfB + 2*j (+e)
// ---------------------------------------------------------------------------
__global__ void pack_weights(const float* __restrict__ w, _Float16* __restrict__ wp,
                             int Cin, int Cout) {
    const int NT  = Cout >> 4;
    const int K   = Cin * 9;
    const int ndw = (K * Cout) >> 1;
    int d = blockIdx.x * blockDim.x + threadIdx.x;
    if (d >= ndw) return;
    int j     = d & 7;
    int lane  = (d >> 3) & 31;
    int nt    = (d >> 8) % NT;
    int chunk = d / (256 * NT);
    int n     = nt * 16 + (lane & 15);
    int halfB = lane >> 4;
    int k0    = chunk * 32 + 16 * halfB + 2 * j;
#pragma unroll
    for (int e = 0; e < 2; ++e) {
        int k   = k0 + e;
        int tap = k / Cin;          // tap-major K
        int c   = k - tap * Cin;
        float v = w[(size_t)(n * Cin + c) * 9 + tap];
        wp[(size_t)d * 2 + e] = (_Float16)v;
    }
}

// ---------------------------------------------------------------------------
// Kernel 2: offset conv (1 -> 18 channels, 3x3, pad 1), f32 out [B,18,28,28].
// ---------------------------------------------------------------------------
__global__ void off_conv(const float* __restrict__ x, const float* __restrict__ w,
                         const float* __restrict__ bias, float* __restrict__ off) {
    int idx = blockIdx.x * blockDim.x + threadIdx.x;     // over B*18*HW
    if (idx >= Bsz * 18 * HW) return;
    int pi = idx % HW;
    int t  = idx / HW;
    int o  = t % 18;
    int b  = t / 18;
    int y = pi / Ws, xq = pi % Ws;
    const float* xb = x + (size_t)b * HW;
    float s = bias[o];
#pragma unroll
    for (int tap = 0; tap < 9; ++tap) {
        int dy = tap / 3 - 1, dx = tap % 3 - 1;
        int yy = y + dy, xx = xq + dx;
        if ((unsigned)yy < (unsigned)Hs && (unsigned)xx < (unsigned)Ws)
            s += xb[yy * Ws + xx] * w[o * 9 + tap];
    }
    off[idx] = s;
}

// ---------------------------------------------------------------------------
// Kernel 3: deformable bilinear sampling + 9->32 einsum + bias + BN1.
// Output NHWC f16: act1[(b*HW + pi)*32 + o]  (contiguous 64B store per pixel).
// ---------------------------------------------------------------------------
__global__ void deform_dcn_bn(const float* __restrict__ x, const float* __restrict__ off,
                              const float* __restrict__ wd, const float* __restrict__ bd,
                              const float* __restrict__ g,  const float* __restrict__ be,
                              const float* __restrict__ mu, const float* __restrict__ va,
                              _Float16* __restrict__ out) {
    int p = blockIdx.x * blockDim.x + threadIdx.x;       // over B*HW
    if (p >= Bsz * HW) return;
    int b = p / HW, pi = p % HW;
    int y = pi / Ws, xq = pi % Ws;
    const float* xb = x + (size_t)b * HW;
    float s9[9];
#pragma unroll
    for (int tap = 0; tap < 9; ++tap) {
        int ky = tap / 3 - 1, kx = tap % 3 - 1;
        float oy = off[((size_t)b * 18 + tap * 2)     * HW + pi];
        float ox = off[((size_t)b * 18 + tap * 2 + 1) * HW + pi];
        float py = (float)y + (float)ky + oy;
        float px = (float)xq + (float)kx + ox;
        float fy0 = floorf(py), fx0 = floorf(px);
        float wy1 = py - fy0, wx1 = px - fx0;
        float wy0 = 1.0f - wy1, wx0 = 1.0f - wx1;
        int iy0 = (int)fy0, ix0 = (int)fx0;
        int iy1 = iy0 + 1,  ix1 = ix0 + 1;
        float acc = 0.0f;
#pragma unroll
        for (int cn = 0; cn < 4; ++cn) {
            int yy = (cn & 2) ? iy1 : iy0;
            int xx = (cn & 1) ? ix1 : ix0;
            float wgt = ((cn & 2) ? wy1 : wy0) * ((cn & 1) ? wx1 : wx0);
            bool valid = (yy >= 0) && (yy <= Hs - 1) && (xx >= 0) && (xx <= Ws - 1);
            int yc = yy < 0 ? 0 : (yy > Hs - 1 ? Hs - 1 : yy);
            int xc = xx < 0 ? 0 : (xx > Ws - 1 ? Ws - 1 : xx);
            float v = xb[yc * Ws + xc];
            acc += (valid ? v : 0.0f) * wgt;
        }
        s9[tap] = acc;
    }
    _Float16* op = out + ((size_t)b * HW + pi) * 32;
    for (int o = 0; o < 32; ++o) {
        float z = bd[o];
#pragma unroll
        for (int tap = 0; tap < 9; ++tap) z += s9[tap] * wd[o * 9 + tap];
        float sc = g[o] * rsqrtf(va[o] + EPSV);
        z = (z - mu[o]) * sc + be[o];
        op[o] = (_Float16)z;
    }
}

// ---------------------------------------------------------------------------
// Kernel 4: implicit-GEMM 3x3 conv via v_wmma_f32_16x16x32_f16, NHWC f16.
// K is tap-major: each 32-wide K chunk has a single (dy,dx) tap, so the whole
// A fragment is two contiguous global_load_b128 per lane (one bounds check).
// Weight fragments for the chunk are staged once per block into LDS with
// global_load_async_to_lds_b128 (ASYNCcnt) and shared by all 8 waves.
// Epilogue: +bias, optional ReLU, inference BN; f16 NHWC store.
// ---------------------------------------------------------------------------
template<int CIN, int NT, bool RELU>
__global__ __launch_bounds__(256)
void conv_wmma(const _Float16* __restrict__ in, const _Float16* __restrict__ wp,
               const float* __restrict__ bias,
               const float* __restrict__ g,  const float* __restrict__ be,
               const float* __restrict__ mu, const float* __restrict__ va,
               _Float16* __restrict__ out) {
    constexpr int Cout   = NT * 16;
    constexpr int CPT    = CIN / 32;         // chunks per tap
    constexpr int CHUNKS = (CIN * 9) / 32;
    __shared__ v16h smem[NT * 32];           // one chunk of B fragments (<= 8KB)

    const int tid  = threadIdx.x;
    const int wave = tid >> 5;
    const int lane = tid & 31;
    const int tile = blockIdx.x * 8 + wave;  // grid exact: tile < TOTAL_TILES

    const int half = lane >> 4;
    const int mn   = lane & 15;
    const int p  = tile * 16 + mn;           // A-row pixel (row M = mn)
    const int b  = p / HW;
    const int pi = p % HW;
    const int y  = pi / Ws;
    const int x  = pi % Ws;

    const v16h* __restrict__ wv = (const v16h*)wp;
    v8f acc[NT] = {};

    for (int kc = 0; kc < CHUNKS; ++kc) {
        __syncthreads();                     // LDS chunk reuse guard
        if (tid < NT * 32) {
            size_t gidx = (size_t)kc * (NT * 32) + tid;
#if __has_builtin(__builtin_amdgcn_global_load_async_to_lds_b128)
            gas_v4i_ptr gp = (gas_v4i_ptr)(wv + gidx);
            las_v4i_ptr lp = (las_v4i_ptr)(&smem[tid]);
            __builtin_amdgcn_global_load_async_to_lds_b128(gp, lp, 0, 0);
            __builtin_amdgcn_global_load_async_to_lds_b128(gp, lp, 16, 0);
#else
            smem[tid] = wv[gidx];
#endif
        }
#if __has_builtin(__builtin_amdgcn_global_load_async_to_lds_b128)
#if __has_builtin(__builtin_amdgcn_s_wait_asynccnt)
        __builtin_amdgcn_s_wait_asynccnt(0);
#else
        asm volatile("s_wait_asynccnt 0" ::: "memory");
#endif
#endif
        __syncthreads();

        // ---- A fragment: tap constant over chunk; channels contiguous (NHWC)
        const int tap   = kc / CPT;
        const int cbase = (kc - tap * CPT) * 32;
        const int dy = tap / 3 - 1;
        const int dx = tap - (tap / 3) * 3 - 1;
        const int yy = y + dy, xx = x + dx;
        v8h lo = {}, hi = {};
        if ((unsigned)yy < (unsigned)Hs && (unsigned)xx < (unsigned)Ws) {
            const _Float16* ap =
                in + ((size_t)b * HW + yy * Ws + xx) * CIN + cbase + 16 * half;
            lo = *(const v8h*)ap;            // k = 16*half + 0..7  (VGPR 0..3)
            hi = *(const v8h*)(ap + 8);      // k = 16*half+8..15   (VGPR 4..7)
        }
        v16h a = __builtin_shufflevector(lo, hi,
                                         0, 1, 2, 3, 4, 5, 6, 7,
                                         8, 9, 10, 11, 12, 13, 14, 15);
#pragma unroll
        for (int nt = 0; nt < NT; ++nt) {
            v16h bf = smem[nt * 32 + lane];  // 2x ds_load_b128
            acc[nt] = __builtin_amdgcn_wmma_f32_16x16x32_f16(
                false, a, false, bf, (short)0, acc[nt], false, false);
        }
    }

    // ---- Epilogue: D layout -> lane n = lane&15, VGPR r holds M = r+8*half
#pragma unroll
    for (int nt = 0; nt < NT; ++nt) {
        const int n = nt * 16 + mn;
        const float sc = g[n] * rsqrtf(va[n] + EPSV);
        const float bi = bias[n], m0 = mu[n], b0 = be[n];
#pragma unroll
        for (int r = 0; r < 8; ++r) {
            const int m  = r + 8 * half;
            const int pr = pi - mn + m;      // same image as p (784 = 49*16)
            float z = acc[nt][r] + bi;
            if (RELU) z = fmaxf(z, 0.0f);
            z = (z - m0) * sc + b0;
            out[((size_t)b * HW + pr) * Cout + n] = (_Float16)z;
        }
    }
}

// ---------------------------------------------------------------------------
// Kernel 5: global mean pool + FC(128->10) + log_softmax. Block per image.
// NHWC input -> pooling reads are fully coalesced across the 128 threads.
// ---------------------------------------------------------------------------
__global__ __launch_bounds__(128)
void pool_fc_lsm(const _Float16* __restrict__ act, const float* __restrict__ wfc,
                 const float* __restrict__ bfc, float* __restrict__ out) {
    __shared__ float pooled[128];
    __shared__ float logits[10];
    const int b = blockIdx.x;
    const int t = threadIdx.x;
    const _Float16* ap = act + (size_t)b * HW * 128 + t;
    float s = 0.0f;
    for (int i = 0; i < HW; ++i) {
        if ((i & 31) == 0) __builtin_prefetch(ap + (size_t)(i + 32) * 128, 0, 0);
        s += (float)ap[(size_t)i * 128];
    }
    pooled[t] = s * (1.0f / (float)HW);
    __syncthreads();
    if (t < 10) {
        float z = bfc[t];
        for (int c = 0; c < 128; ++c) z += pooled[c] * wfc[t * 128 + c];
        logits[t] = z;
    }
    __syncthreads();
    if (t < 10) {
        float mx = logits[0];
#pragma unroll
        for (int j = 1; j < 10; ++j) mx = fmaxf(mx, logits[j]);
        float se = 0.0f;
#pragma unroll
        for (int j = 0; j < 10; ++j) se += expf(logits[j] - mx);
        out[b * 10 + t] = logits[t] - mx - logf(se);
    }
}

// ---------------------------------------------------------------------------
extern "C" void kernel_launch(void* const* d_in, const int* in_sizes, int n_in,
                              void* d_out, int out_size, void* d_ws, size_t ws_size,
                              hipStream_t stream) {
    (void)in_sizes; (void)n_in; (void)out_size; (void)ws_size;
    // setup_inputs() dict order:
    const float* x     = (const float*)d_in[0];
    const float* w_off = (const float*)d_in[1];
    const float* b_off = (const float*)d_in[2];
    const float* w_dcn = (const float*)d_in[3];
    const float* b_dcn = (const float*)d_in[4];
    const float* w2    = (const float*)d_in[5];
    const float* b2    = (const float*)d_in[6];
    const float* w3    = (const float*)d_in[7];
    const float* b3    = (const float*)d_in[8];
    const float* w4    = (const float*)d_in[9];
    const float* b4    = (const float*)d_in[10];
    const float* w_fc  = (const float*)d_in[11];
    const float* b_fc  = (const float*)d_in[12];
    const float* bn1g = (const float*)d_in[13]; const float* bn1b = (const float*)d_in[14];
    const float* bn1m = (const float*)d_in[15]; const float* bn1v = (const float*)d_in[16];
    const float* bn2g = (const float*)d_in[17]; const float* bn2b = (const float*)d_in[18];
    const float* bn2m = (const float*)d_in[19]; const float* bn2v = (const float*)d_in[20];
    const float* bn3g = (const float*)d_in[21]; const float* bn3b = (const float*)d_in[22];
    const float* bn3m = (const float*)d_in[23]; const float* bn3v = (const float*)d_in[24];
    const float* bn4g = (const float*)d_in[25]; const float* bn4b = (const float*)d_in[26];
    const float* bn4m = (const float*)d_in[27]; const float* bn4v = (const float*)d_in[28];

    // Workspace layout (all offsets 256B aligned):
    //   region A : 1024*784*128 f16 = 205,520,896 B  (act ping, NHWC)
    //   region B : same size                         (off f32 / act pong, NHWC)
    //   packed weights for conv2/conv3/conv4
    const size_t A_BYTES = (size_t)Bsz * HW * 128 * sizeof(_Float16); // 205,520,896
    char* base = (char*)d_ws;
    _Float16* actA  = (_Float16*)(base);
    _Float16* actB  = (_Float16*)(base + A_BYTES);
    float*    offB  = (float*)   (base + A_BYTES);        // lives until conv2 writes actB
    _Float16* wp2   = (_Float16*)(base + 2 * A_BYTES);                    // 288*64  halves
    _Float16* wp3   = (_Float16*)(base + 2 * A_BYTES + 36864);            // 576*128 halves
    _Float16* wp4   = (_Float16*)(base + 2 * A_BYTES + 36864 + 147456);   // 1152*128 halves

    // 1) pack weights into WMMA B-fragment order (f16, tap-major K)
    {
        int dw2 = (32 * 9 * 64) / 2;     // 9216
        int dw3 = (64 * 9 * 128) / 2;    // 36864
        int dw4 = (128 * 9 * 128) / 2;   // 73728
        pack_weights<<<(dw2 + 255) / 256, 256, 0, stream>>>(w2, wp2, 32, 64);
        pack_weights<<<(dw3 + 255) / 256, 256, 0, stream>>>(w3, wp3, 64, 128);
        pack_weights<<<(dw4 + 255) / 256, 256, 0, stream>>>(w4, wp4, 128, 128);
    }

    // 2) offset conv -> offB (f32, [B,18,28,28])
    {
        int n = Bsz * 18 * HW;
        off_conv<<<(n + 255) / 256, 256, 0, stream>>>(x, w_off, b_off, offB);
    }

    // 3) deformable sampling + dcn einsum + bias + BN1 -> actA (NHWC f16, C=32)
    {
        int n = Bsz * HW;
        deform_dcn_bn<<<(n + 255) / 256, 256, 0, stream>>>(
            x, offB, w_dcn, b_dcn, bn1g, bn1b, bn1m, bn1v, actA);
    }

    // 4) WMMA conv stack: 8 waves/block, one 16-pixel tile per wave
    const int blocks = TOTAL_TILES / 8;   // 6272
    conv_wmma<32, 4, true><<<blocks, 256, 0, stream>>>(
        actA, wp2, b2, bn2g, bn2b, bn2m, bn2v, actB);    // 32 -> 64
    conv_wmma<64, 8, true><<<blocks, 256, 0, stream>>>(
        actB, wp3, b3, bn3g, bn3b, bn3m, bn3v, actA);    // 64 -> 128
    conv_wmma<128, 8, true><<<blocks, 256, 0, stream>>>(
        actA, wp4, b4, bn4g, bn4b, bn4m, bn4v, actB);    // 128 -> 128

    // 5) pool + FC + log_softmax -> d_out ([1024,10] f32)
    pool_fc_lsm<<<Bsz, 128, 0, stream>>>(actB, w_fc, b_fc, (float*)d_out);
}
